// SimpleTransformer_60825326846283
// MI455X (gfx1250) — compile-verified, matrix operand order checked by reading, not run on hardware
//
#include <hip/hip_runtime.h>

// ---------------------------------------------------------------------------
// SimpleTransformer on MI455X (gfx1250): bf16 WMMA pipeline, flash attention,
// GEGLU fused into the W1 GEMM epilogue, async global->LDS staging
// (GLOBAL_LOAD_ASYNC_TO_LDS_B128 / ASYNCcnt). ~98 MB workspace, fits in L2.
// ---------------------------------------------------------------------------

typedef __attribute__((ext_vector_type(16))) __bf16       v16bf;
typedef __attribute__((ext_vector_type(8)))  float        v8f;
typedef __attribute__((ext_vector_type(4)))  unsigned int u32x4;
typedef __attribute__((ext_vector_type(4)))  int          i32x4;
typedef __attribute__((ext_vector_type(4)))  float        f32x4;

using bf16 = unsigned short;   // raw bf16 storage

union FragAB { v16bf v; u32x4 q[2]; };

// --- async global->LDS (GLOBAL_LOAD_ASYNC_TO_LDS_B128, ASYNCcnt) -----------
#if defined(__gfx1250__) && __has_builtin(__builtin_amdgcn_global_load_async_to_lds_b128)
#define HAVE_ASYNC_LDS 1
__device__ __forceinline__ void async_cp16(void* dst_lds, void* src_glob) {
    __builtin_amdgcn_global_load_async_to_lds_b128(
        (__attribute__((address_space(1))) i32x4*)src_glob,
        (__attribute__((address_space(3))) i32x4*)dst_lds, 0, 0);
}
#define ASYNC_WAIT() asm volatile("s_wait_asynccnt 0x0" ::: "memory")
#else
#define HAVE_ASYNC_LDS 0
#define ASYNC_WAIT()
#endif

__device__ __forceinline__ bf16 f2bf(float x) {
    unsigned int u = __float_as_uint(x);
    u = (u + 0x7FFFu + ((u >> 16) & 1u)) >> 16;   // round-to-nearest-even
    return (bf16)u;
}
__device__ __forceinline__ float bf2f(bf16 h) {
    return __uint_as_float(((unsigned int)h) << 16);
}
__device__ __forceinline__ float gelu_exact(float g) {
    return 0.5f * g * (1.f + erff(g * 0.70710678118654752f));
}

// Load one 16x32(bf16) A-fragment (or its dual B-fragment) from a row-major
// tile: lane l<16 -> row l, K in {0..7, 16..23}; lane l>=16 -> row l-16,
// K in {8..15, 24..31}. Two 16B loads per lane. (ISA 7.12.2 layout.)
__device__ __forceinline__ void ld_frag(const bf16* base, int stride, int lane,
                                        FragAB& f) {
    const int r  = lane & 15;
    const int kb = (lane < 16) ? 0 : 8;
    const bf16* p = base + r * stride + kb;
    f.q[0] = *(const u32x4*)(p);
    f.q[1] = *(const u32x4*)(p + 16);
}

// ---------------------------------------------------------------------------
// Shared GEMM core pieces: 256 threads, block tile 128x128x32, 8 waves of
// 64x32 (4x2 WMMA tiles), double-buffered LDS, async staging + prefetch.
// ---------------------------------------------------------------------------
#if HAVE_ASYNC_LDS
#define STAGE_CP(ldst, gsrc) async_cp16((void*)(ldst), (void*)(gsrc))
#else
#define STAGE_CP(ldst, gsrc) *(u32x4*)(ldst) = *(const u32x4*)(gsrc)
#endif

#define GEMM_PROLOGUE()                                                        \
    __shared__ __align__(16) bf16 ldsA[2][128 * 32];                           \
    __shared__ __align__(16) bf16 ldsB[2][128 * 32];                           \
    const int tid  = threadIdx.x;                                              \
    const int lane = tid & 31;                                                 \
    const int wave = tid >> 5;                                                 \
    const int mBlk = blockIdx.x * 128;                                         \
    const int nBlk = blockIdx.y * 128;                                         \
    const int wm   = (wave >> 2) * 64;                                         \
    const int wn   = (wave & 3) * 32;                                          \
    v8f acc[4][2];                                                             \
    _Pragma("unroll") for (int i = 0; i < 4; ++i)                              \
        _Pragma("unroll") for (int j = 0; j < 2; ++j) acc[i][j] = v8f{};       \
    auto stage = [&](int kt, int buf, int nk) {                                \
        const int row = tid >> 1;                                              \
        const int seg = (tid & 1) * 16;                                        \
        const bf16* ga = A  + (size_t)(mBlk + row) * K + kt * 32 + seg;        \
        const bf16* gb = Bt + (size_t)(nBlk + row) * K + kt * 32 + seg;        \
        STAGE_CP(&ldsA[buf][row * 32 + seg],     ga);                          \
        STAGE_CP(&ldsA[buf][row * 32 + seg + 8], ga + 8);                      \
        STAGE_CP(&ldsB[buf][row * 32 + seg],     gb);                          \
        STAGE_CP(&ldsB[buf][row * 32 + seg + 8], gb + 8);                      \
        if (kt + 1 < nk) {     /* global_prefetch_b8 one k-tile further */     \
            __builtin_prefetch(ga + 32, 0, 1);                                 \
            __builtin_prefetch(gb + 32, 0, 1);                                 \
        }                                                                      \
    };                                                                         \
    const int nk = K >> 5;                                                     \
    stage(0, 0, nk);                                                           \
    ASYNC_WAIT();                                                              \
    __syncthreads();                                                           \
    for (int kt = 0; kt < nk; ++kt) {                                          \
        const int cur = kt & 1;                                                \
        if (kt + 1 < nk) stage(kt + 1, cur ^ 1, nk);                           \
        FragAB af[4], bfr[2];                                                  \
        _Pragma("unroll") for (int i = 0; i < 4; ++i)                          \
            ld_frag(&ldsA[cur][(wm + i * 16) * 32], 32, lane, af[i]);          \
        _Pragma("unroll") for (int j = 0; j < 2; ++j)                          \
            ld_frag(&ldsB[cur][(wn + j * 16) * 32], 32, lane, bfr[j]);         \
        _Pragma("unroll") for (int i = 0; i < 4; ++i)                          \
            _Pragma("unroll") for (int j = 0; j < 2; ++j)                      \
                acc[i][j] = __builtin_amdgcn_wmma_f32_16x16x32_bf16(           \
                    false, af[i].v, false, bfr[j].v, (short)0, acc[i][j],      \
                    false, false);                                             \
        ASYNC_WAIT();                                                          \
        __syncthreads();                                                       \
    }

// C[M,N] = A * Bt^T (+bias) (+res), f32 or bf16(*oscale) output.
template <bool BF16OUT>
__global__ __launch_bounds__(256) void gemm_wmma_bf16(
    const bf16* __restrict__ A, const bf16* __restrict__ Bt,
    const float* __restrict__ bias, const float* __restrict__ res,
    float* __restrict__ Cf, bf16* __restrict__ Cb, int M, int N, int K,
    float oscale) {
    GEMM_PROLOGUE()
#pragma unroll
    for (int i = 0; i < 4; ++i) {
#pragma unroll
        for (int j = 0; j < 2; ++j) {
            const int col = nBlk + wn + j * 16 + (lane & 15);
            const float bv = bias ? bias[col] : 0.f;
#pragma unroll
            for (int r = 0; r < 8; ++r) {
                const int row = mBlk + wm + i * 16 + r + ((lane < 16) ? 0 : 8);
                float v = acc[i][j][r] + bv;
                if constexpr (BF16OUT) {
                    Cb[(size_t)row * N + col] = f2bf(v * oscale);
                } else {
                    if (res) v += res[(size_t)row * N + col];
                    Cf[(size_t)row * N + col] = v;
                }
            }
        }
    }
}

// GEGLU-fused W1 GEMM: Bt is the *paired-permuted* W1 [4096][512] where
// N-block rows nb*32+{0..15} hold the `a` columns nb*16+{0..15} and rows
// nb*32+{16..31} hold the matching `g` columns 2048+nb*16+{0..15}. Each
// wave's two 16-wide N sub-tiles are then exactly (a, g) for the same output
// column -> p = (a+ba)*gelu(g+bg) is per-lane math. Output P [M][2048] bf16.
__global__ __launch_bounds__(256) void gemm_wmma_geglu(
    const bf16* __restrict__ A, const bf16* __restrict__ Bt,
    const float* __restrict__ b1, bf16* __restrict__ P, int M, int K) {
    GEMM_PROLOGUE()
    const int nb = (nBlk + wn) >> 5;            // 16-col pair-block index
    const int c  = nb * 16 + (lane & 15);       // output column in [0,2048)
    const float ba = b1[c];
    const float bg = b1[2048 + c];
#pragma unroll
    for (int i = 0; i < 4; ++i) {
#pragma unroll
        for (int r = 0; r < 8; ++r) {
            const int row = mBlk + wm + i * 16 + r + ((lane < 16) ? 0 : 8);
            const float av = acc[i][0][r] + ba;
            const float gv = acc[i][1][r] + bg;
            P[(size_t)row * 2048 + c] = f2bf(av * gelu_exact(gv));
        }
    }
}

// ---------------------------------------------------------------------------
// Flash attention. B=8, N=1024, H=8, Dh=64, row stride 512 (packed heads).
// Q arrives pre-scaled by D_HEAD^-0.5 (folded into the Wq GEMM epilogue).
// 128 threads = 4 waves; each wave owns 16 query rows; 64-row query tile per
// block; streams 16 key tiles of 64. grid = (B*H, N/64).
// ---------------------------------------------------------------------------
__global__ __launch_bounds__(128) void flash_attn_wmma(
    const bf16* __restrict__ Q, const bf16* __restrict__ Kb,
    const bf16* __restrict__ Vb, bf16* __restrict__ O) {
    __shared__ __align__(16) bf16 ldsK[64 * 64];
    __shared__ __align__(16) bf16 ldsVt[64 * 64];
    __shared__ __align__(16) bf16 ldsP[4][16 * 64];

    const int tid = threadIdx.x, lane = tid & 31, wave = tid >> 5;
    const int b = blockIdx.x >> 3, h = blockIdx.x & 7;
    const int qt = blockIdx.y;
    const int rowBase = b * 1024 + qt * 64;
    const int h0 = h * 64;
    const int D = 512;

    // Q fragments (16x64 per wave -> two 16x32 A-frags), read straight from L2
    FragAB qf[2];
    {
        const int r  = lane & 15;
        const int kb = (lane < 16) ? 0 : 8;
        const bf16* qp = Q + (size_t)(rowBase + wave * 16 + r) * D + h0;
#pragma unroll
        for (int kk = 0; kk < 2; ++kk) {
            const bf16* p = qp + kk * 32 + kb;
            qf[kk].q[0] = *(const u32x4*)(p);
            qf[kk].q[1] = *(const u32x4*)(p + 16);
        }
    }

    float mrow[8], lrow[8];
    v8f oacc[4];
#pragma unroll
    for (int r = 0; r < 8; ++r) { mrow[r] = -1e30f; lrow[r] = 0.f; }
#pragma unroll
    for (int t = 0; t < 4; ++t) oacc[t] = v8f{};

    for (int kt = 0; kt < 16; ++kt) {
        __syncthreads();
        {   // stage K tile row-major (async DMA) + V tile transposed (manual)
            const int row = tid >> 1;
            const int seg = (tid & 1) * 32;
            const bf16* gk = Kb + (size_t)(b * 1024 + kt * 64 + row) * D + h0 + seg;
            const bf16* gv = Vb + (size_t)(b * 1024 + kt * 64 + row) * D + h0 + seg;
#pragma unroll
            for (int u = 0; u < 4; ++u)
                STAGE_CP(&ldsK[row * 64 + seg + u * 8], gk + u * 8);
            __align__(16) bf16 tmp[32];
#pragma unroll
            for (int u = 0; u < 4; ++u)
                *(u32x4*)&tmp[u * 8] = *(const u32x4*)(gv + u * 8);
#pragma unroll
            for (int u = 0; u < 32; ++u)
                ldsVt[(seg + u) * 64 + row] = tmp[u];
        }
        ASYNC_WAIT();
        __syncthreads();

        // S = Q K^T  (4 key sub-tiles of 16, K-dim = Dh = 64 -> 2 steps)
        v8f s[4];
#pragma unroll
        for (int t = 0; t < 4; ++t) {
            s[t] = v8f{};
#pragma unroll
            for (int kk = 0; kk < 2; ++kk) {
                FragAB kf;
                ld_frag(&ldsK[(t * 16) * 64 + kk * 32], 64, lane, kf);
                s[t] = __builtin_amdgcn_wmma_f32_16x16x32_bf16(
                    false, qf[kk].v, false, kf.v, (short)0, s[t], false, false);
            }
        }

        // online softmax: row r lives in reg r across its 16-lane group
#pragma unroll
        for (int r = 0; r < 8; ++r) {
            float mx = fmaxf(fmaxf(s[0][r], s[1][r]), fmaxf(s[2][r], s[3][r]));
#pragma unroll
            for (int off = 1; off < 16; off <<= 1)
                mx = fmaxf(mx, __shfl_xor(mx, off, 32));
            const float mnew  = fmaxf(mrow[r], mx);
            const float alpha = __expf(mrow[r] - mnew);
            mrow[r] = mnew;
            float psum = 0.f;
#pragma unroll
            for (int t = 0; t < 4; ++t) {
                const float p = __expf(s[t][r] - mnew);
                s[t][r] = p;
                psum += p;
            }
#pragma unroll
            for (int off = 1; off < 16; off <<= 1)
                psum += __shfl_xor(psum, off, 32);
            lrow[r] = lrow[r] * alpha + psum;
#pragma unroll
            for (int t = 0; t < 4; ++t) oacc[t][r] *= alpha;
        }

        // P: C-layout regs -> A-layout via wave-private LDS patch
        bf16* pp = ldsP[wave];
#pragma unroll
        for (int t = 0; t < 4; ++t)
#pragma unroll
            for (int r = 0; r < 8; ++r) {
                const int prow = r + ((lane < 16) ? 0 : 8);
                pp[prow * 64 + t * 16 + (lane & 15)] = f2bf(s[t][r]);
            }
        asm volatile("s_wait_dscnt 0x0" ::: "memory");

        FragAB pf[2];
#pragma unroll
        for (int kk = 0; kk < 2; ++kk) ld_frag(pp + kk * 32, 64, lane, pf[kk]);
#pragma unroll
        for (int dt = 0; dt < 4; ++dt)
#pragma unroll
            for (int kk = 0; kk < 2; ++kk) {
                FragAB vf;
                ld_frag(&ldsVt[(dt * 16) * 64 + kk * 32], 64, lane, vf);
                oacc[dt] = __builtin_amdgcn_wmma_f32_16x16x32_bf16(
                    false, pf[kk].v, false, vf.v, (short)0, oacc[dt], false, false);
            }
    }

#pragma unroll
    for (int dt = 0; dt < 4; ++dt)
#pragma unroll
        for (int r = 0; r < 8; ++r) {
            const int row = rowBase + wave * 16 + r + ((lane < 16) ? 0 : 8);
            const int col = h0 + dt * 16 + (lane & 15);
            O[(size_t)row * 512 + col] = f2bf(oacc[dt][r] / lrow[r]);
        }
}

// ---------------------------------------------------------------------------
// LayerNorm over D=512: one row per wave32, 16 elems/lane, shfl reductions.
// ---------------------------------------------------------------------------
__global__ __launch_bounds__(256) void layernorm_bf16(
    const float* __restrict__ X, const float* __restrict__ w,
    const float* __restrict__ bia, bf16* __restrict__ Y) {
    const int lane = threadIdx.x & 31, wave = threadIdx.x >> 5;
    const int row = blockIdx.x * 8 + wave;
    const float* xp = X + (size_t)row * 512 + lane * 16;
    f32x4 a[4];
#pragma unroll
    for (int u = 0; u < 4; ++u) a[u] = *(const f32x4*)(xp + u * 4);
    float s = 0.f, ss = 0.f;
#pragma unroll
    for (int u = 0; u < 4; ++u)
#pragma unroll
        for (int e = 0; e < 4; ++e) { const float v = a[u][e]; s += v; ss += v * v; }
#pragma unroll
    for (int off = 1; off < 32; off <<= 1) {
        s  += __shfl_xor(s, off, 32);
        ss += __shfl_xor(ss, off, 32);
    }
    const float mean = s * (1.f / 512.f);
    const float rstd = rsqrtf(ss * (1.f / 512.f) - mean * mean + 1e-5f);
    const float* wp = w + lane * 16;
    const float* bp = bia + lane * 16;
    __align__(16) bf16 ob[16];
#pragma unroll
    for (int u = 0; u < 4; ++u) {
        const f32x4 wv = *(const f32x4*)(wp + u * 4);
        const f32x4 bv = *(const f32x4*)(bp + u * 4);
#pragma unroll
        for (int e = 0; e < 4; ++e)
            ob[u * 4 + e] = f2bf((a[u][e] - mean) * rstd * wv[e] + bv[e]);
    }
    u32x4* dst = (u32x4*)(Y + (size_t)row * 512 + lane * 16);
    dst[0] = *(const u32x4*)&ob[0];
    dst[1] = *(const u32x4*)&ob[8];
}

// ---------------------------------------------------------------------------
// Weight conversion kernels
// ---------------------------------------------------------------------------
__global__ void cvt_w_transpose(const float* __restrict__ W,
                                bf16* __restrict__ Wt, int K, int N) {
    const int i = blockIdx.x * 256 + threadIdx.x;
    if (i >= K * N) return;
    const int k = i / N, n = i % N;
    Wt[(size_t)n * K + k] = f2bf(W[i]);
}

// W1 [512][4096] -> paired-permuted transposed [4096][512] for the GEGLU GEMM.
__global__ void cvt_w1_geglu(const float* __restrict__ W,
                             bf16* __restrict__ Wt, int K) {
    const int i = blockIdx.x * 256 + threadIdx.x;
    if (i >= K * 4096) return;
    const int k = i / 4096, n = i % 4096;
    const int c  = (n < 2048) ? n : (n - 2048);
    const int nb = c >> 4;
    const int pos = nb * 32 + ((n < 2048) ? 0 : 16) + (c & 15);
    Wt[(size_t)pos * K + k] = f2bf(W[i]);
}

__global__ void cvt_f32_to_bf16(const float* __restrict__ X,
                                bf16* __restrict__ Y, int n) {
    const int i = blockIdx.x * 256 + threadIdx.x;
    if (i < n) Y[i] = f2bf(X[i]);
}

// ---------------------------------------------------------------------------
// Host pipeline
// ---------------------------------------------------------------------------
extern "C" void kernel_launch(void* const* d_in, const int* in_sizes, int n_in,
                              void* d_out, int out_size, void* d_ws,
                              size_t ws_size, hipStream_t stream) {
    const float* x      = (const float*)d_in[0];
    const float* ln0_w  = (const float*)d_in[1];
    const float* ln0_b  = (const float*)d_in[2];
    const float* pin_w  = (const float*)d_in[3];
    const float* pin_b  = (const float*)d_in[4];
    const float* Wq     = (const float*)d_in[5];
    const float* Wk     = (const float*)d_in[6];
    const float* Wv     = (const float*)d_in[7];
    const float* Wo     = (const float*)d_in[8];
    const float* bo     = (const float*)d_in[9];
    const float* ln1_w  = (const float*)d_in[10];
    const float* ln1_b  = (const float*)d_in[11];
    const float* ln2_w  = (const float*)d_in[12];
    const float* ln2_b  = (const float*)d_in[13];
    const float* W1     = (const float*)d_in[14];
    const float* b1     = (const float*)d_in[15];
    const float* W2     = (const float*)d_in[16];
    const float* b2     = (const float*)d_in[17];
    const float* pout_w = (const float*)d_in[18];
    const float* pout_b = (const float*)d_in[19];
    float* out = (float*)d_out;

    const int M = 8 * 1024;   // 8192 tokens
    const int D = 512;

    char* ws = (char*)d_ws;
    size_t off = 0;
    auto alloc = [&](size_t bytes) -> char* {
        off = (off + 255) & ~(size_t)255;
        char* p = ws + off;
        off += bytes;
        return p;
    };

    bf16* pinT  = (bf16*)alloc((size_t)512 * 512 * 2);
    bf16* poutT = (bf16*)alloc((size_t)512 * 512 * 2);
    bf16 *WqT[2], *WkT[2], *WvT[2], *WoT[2], *W1P[2], *W2T[2];
    for (int l = 0; l < 2; ++l) {
        WqT[l] = (bf16*)alloc((size_t)512 * 512 * 2);
        WkT[l] = (bf16*)alloc((size_t)512 * 512 * 2);
        WvT[l] = (bf16*)alloc((size_t)512 * 512 * 2);
        WoT[l] = (bf16*)alloc((size_t)512 * 512 * 2);
        W1P[l] = (bf16*)alloc((size_t)4096 * 512 * 2);   // paired-permuted
        W2T[l] = (bf16*)alloc((size_t)512 * 2048 * 2);   // [N=512][K=2048]
    }
    bf16*  lnbuf = (bf16*)alloc((size_t)M * 512 * 2);
    bf16*  qbuf  = (bf16*)alloc((size_t)M * 512 * 2);
    bf16*  kbuf  = (bf16*)alloc((size_t)M * 512 * 2);
    bf16*  vbuf  = (bf16*)alloc((size_t)M * 512 * 2);
    bf16*  pbuf  = (bf16*)alloc((size_t)M * 2048 * 2);
    float* ybuf  = (float*)alloc((size_t)M * 512 * 4);

    auto cvtw = [&](const float* W, bf16* Wt, int K, int N) {
        const int n = K * N;
        cvt_w_transpose<<<(n + 255) / 256, 256, 0, stream>>>(W, Wt, K, N);
    };
    auto gemm_f32 = [&](const bf16* A, const bf16* Bt, const float* bias,
                        const float* res, float* C, int m, int n, int k) {
        gemm_wmma_bf16<false><<<dim3(m / 128, n / 128), 256, 0, stream>>>(
            A, Bt, bias, res, C, nullptr, m, n, k, 1.f);
    };
    auto gemm_b16 = [&](const bf16* A, const bf16* Bt, const float* bias,
                        bf16* C, int m, int n, int k, float oscale) {
        gemm_wmma_bf16<true><<<dim3(m / 128, n / 128), 256, 0, stream>>>(
            A, Bt, bias, nullptr, nullptr, C, m, n, k, oscale);
    };
    auto ln = [&](const float* X, const float* w, const float* b, bf16* Y) {
        layernorm_bf16<<<M / 8, 256, 0, stream>>>(X, w, b, Y);
    };

    // weight precision conversion (+ transpose to [N][K])
    cvtw(pin_w, pinT, 512, 512);
    cvtw(pout_w, poutT, 512, 512);
    for (int l = 0; l < 2; ++l) {
        cvtw(Wq + (size_t)l * 512 * 512, WqT[l], 512, 512);
        cvtw(Wk + (size_t)l * 512 * 512, WkT[l], 512, 512);
        cvtw(Wv + (size_t)l * 512 * 512, WvT[l], 512, 512);
        cvtw(Wo + (size_t)l * 512 * 512, WoT[l], 512, 512);
        cvt_w1_geglu<<<(512 * 4096 + 255) / 256, 256, 0, stream>>>(
            W1 + (size_t)l * 512 * 4096, W1P[l], 512);
        cvtw(W2 + (size_t)l * 2048 * 512, W2T[l], 2048, 512);
    }

    // y = LN(x) @ pin + pin_b
    ln(x, ln0_w, ln0_b, lnbuf);
    gemm_f32(lnbuf, pinT, pin_b, nullptr, ybuf, M, D, D);

    for (int l = 0; l < 2; ++l) {
        // attention block (Q pre-scaled by D_HEAD^-0.5 in the GEMM epilogue)
        ln(ybuf, ln1_w + l * 512, ln1_b + l * 512, lnbuf);
        gemm_b16(lnbuf, WqT[l], nullptr, qbuf, M, D, D, 0.125f);
        gemm_b16(lnbuf, WkT[l], nullptr, kbuf, M, D, D, 1.f);
        gemm_b16(lnbuf, WvT[l], nullptr, vbuf, M, D, D, 1.f);
        flash_attn_wmma<<<dim3(64, 16), 128, 0, stream>>>(qbuf, kbuf, vbuf,
                                                          lnbuf);
        gemm_f32(lnbuf, WoT[l], bo + l * 512, ybuf, ybuf, M, D, D);

        // GEGLU feed-forward, gate fused into the W1 GEMM epilogue
        ln(ybuf, ln2_w + l * 512, ln2_b + l * 512, lnbuf);
        gemm_wmma_geglu<<<dim3(M / 128, 4096 / 128), 256, 0, stream>>>(
            lnbuf, W1P[l], b1 + (size_t)l * 4096, pbuf, M, D);
        gemm_f32(pbuf, W2T[l], b2 + l * 512, ybuf, ybuf, M, D, 2048);
    }

    // out = y @ pout + pout_b
    cvt_f32_to_bf16<<<(M * D) / 256, 256, 0, stream>>>(ybuf, lnbuf, M * D);
    gemm_f32(lnbuf, poutT, pout_b, nullptr, out, M, D, D);
}
